// LFMMILoss_36361193128162
// MI455X (gfx1250) — compile-verified, matrix operand order checked by reading, not run on hardware
//
#include <hip/hip_runtime.h>
#include <hip/hip_bf16.h>
#include <math.h>

// ---------------------------------------------------------------------------
// LF-MMI loss on MI455X (gfx1250).
// Forward algorithm per step as a 16 x S x S f16 WMMA GEMM in the scaled-prob
// domain:  alpha_new[b,j] = m_b + log( (exp(alpha-m) @ exp(trans))[b,j] ) + e.
// One workgroup (16 wave32s) per FSM; alpha + exp(alpha) tiles in LDS;
// exp(trans) pre-packed into the CDNA5 WMMA B-operand layout in workspace.
// ---------------------------------------------------------------------------

typedef __attribute__((ext_vector_type(16))) _Float16 v16h;
typedef __attribute__((ext_vector_type(8)))  _Float16 v8h;
typedef __attribute__((ext_vector_type(8)))  float    v8f;

#define BATCH 16
#define TLEN  512
#define PDIM  2048

__device__ __forceinline__ float wave_max32(float m) {
#pragma unroll
  for (int off = 16; off > 0; off >>= 1) m = fmaxf(m, __shfl_xor(m, off, 32));
  return m;
}
__device__ __forceinline__ float wave_sum32(float s) {
#pragma unroll
  for (int off = 16; off > 0; off >>= 1) s += __shfl_xor(s, off, 32);
  return s;
}

// K offset inside a 32-wide K-chunk for (vgpr v, packed elem e, lane-half h)
// per the CDNA5 16-bit operand VGPR layout (CDNA5 ISA §7.12.2).
__device__ __forceinline__ int wmma_koff(int v, int e, int half) {
  return (v < 4 ? 2 * v : 16 + 2 * (v - 4)) + half * 8 + e;
}

// Pack exp(trans[S][S]) (row-major, [from][to]) into the WMMA B-operand
// layout so a lane's full 16-f16 fragment is one contiguous 32B load:
//   out[ ((kc*S + col)*2 + half)*16 + (2v+e) ] = exp(trans[kc*32+koff][col])
template <int S, int LOG2S>
__global__ void lfmmi_pack_trans(const float* __restrict__ trans,
                                 _Float16* __restrict__ out) {
  int o = blockIdx.x * blockDim.x + threadIdx.x;
  if (o >= S * S) return;
  int idx16 = o & 15;
  int v = idx16 >> 1, e = idx16 & 1;
  int half = (o >> 4) & 1;
  int col  = (o >> 5) & (S - 1);
  int kc   = o >> (5 + LOG2S);
  int k = kc * 32 + wmma_koff(v, e, half);
  out[o] = (_Float16)__expf(trans[(size_t)k * S + col]);
}

// One block, 512 threads (16 wave32s). NT = S/16 column tiles, TPW per wave.
template <int S, int LOG2S, int TPW>
__global__ __launch_bounds__(512) void lfmmi_forward(
    const float* __restrict__ input,      // [B, T, P]
    const int* __restrict__ seqlen,       // [B]
    const float* __restrict__ init,       // [S]
    const _Float16* __restrict__ Bpack,   // packed exp(trans)
    const float* __restrict__ finals,     // [S]
    const int* __restrict__ pdf,          // [S]
    float* __restrict__ out_llh)          // [1] : sum_b logprob
{
  extern __shared__ char smem[];
  float*    alpha = (float*)smem;                         // 16*S f32
  _Float16* aprob = (_Float16*)(smem + BATCH * S * 4);    // 16*S f16 (WMMA A)
  float*    mrow  = (float*)(smem + BATCH * S * 6);       // 16 row maxima
  float*    redb  = mrow + BATCH;                         // 16 per-row llh

  const int tid  = threadIdx.x;
  const int wave = tid >> 5;
  const int lane = tid & 31;
  const int half = lane >> 4;
  const int l15  = lane & 15;

  // Hoist per-lane constants: pdf ids of owned columns, seqlens of owned rows.
  int pdfreg[TPW];
#pragma unroll
  for (int t = 0; t < TPW; ++t)
    pdfreg[t] = pdf[(wave * TPW + t) * 16 + l15];
  int slen[8];
#pragma unroll
  for (int v = 0; v < 8; ++v) slen[v] = seqlen[half * 8 + v];

  // t = 0 : alpha0[b,j] = init[j] + input[b,0,pdf[j]]
  for (int idx = tid; idx < BATCH * S; idx += blockDim.x) {
    int b = idx >> LOG2S, j = idx & (S - 1);
    alpha[idx] = init[j] + input[(size_t)b * TLEN * PDIM + pdf[j]];
  }
  __syncthreads();

  for (int tstep = 1; tstep < TLEN; ++tstep) {
    // --- phase A1: per-batch-row max (one wave per row) ------------------
    {
      float m = -3.402823466e38f;
      for (int j = lane; j < S; j += 32)
        m = fmaxf(m, alpha[(wave << LOG2S) + j]);
      m = wave_max32(m);
      if (lane == 0) mrow[wave] = m;
    }
    __syncthreads();
    // --- phase A2: scaled probabilities -> f16 WMMA A tile in LDS --------
    for (int idx = tid; idx < BATCH * S; idx += blockDim.x)
      aprob[idx] = (_Float16)__expf(alpha[idx] - mrow[idx >> LOG2S]);
    __syncthreads();

    // --- phase B: 16 x S x S GEMM on the matrix pipes --------------------
    v8f acc[TPW];
#pragma unroll
    for (int t = 0; t < TPW; ++t) acc[t] = (v8f)0.0f;

    for (int kc = 0; kc < S / 32; ++kc) {
      union { v16h v; v8h h[2]; } af;
      const _Float16* ap = aprob + (l15 << LOG2S) + kc * 32 + half * 8;
      af.h[0] = *(const v8h*)(ap);        // K run [h*8, h*8+8)
      af.h[1] = *(const v8h*)(ap + 16);   // K run [16+h*8, 16+h*8+8)
#pragma unroll
      for (int t = 0; t < TPW; ++t) {
        int col = (wave * TPW + t) * 16 + l15;
        v16h bf = *(const v16h*)(Bpack + ((size_t)(kc * S + col) * 2 + half) * 16);
        acc[t] = __builtin_amdgcn_wmma_f32_16x16x32_f16(
            false, af.v, false, bf, (short)0, acc[t], false, false);
      }
    }

    // --- phase C: back to log domain + emission + seqlen mask ------------
#pragma unroll
    for (int t = 0; t < TPW; ++t) {
      int j = (wave * TPW + t) * 16 + l15;
#pragma unroll
      for (int v = 0; v < 8; ++v) {
        int b = half * 8 + v;                 // D layout: row = v + 8*(lane>=16)
        if (tstep < slen[v]) {
          float e = input[((size_t)b * TLEN + tstep) * PDIM + pdfreg[t]];
          alpha[(b << LOG2S) + j] = mrow[b] + __logf(acc[t][v]) + e;
        }
      }
    }
    __syncthreads();
  }

  // Final: llh[b] = logsumexp_j(alpha[b,j] + final[j]); one wave per row.
  {
    int b = wave;
    float m = -3.402823466e38f;
    for (int j = lane; j < S; j += 32)
      m = fmaxf(m, alpha[(b << LOG2S) + j] + finals[j]);
    m = wave_max32(m);
    float s = 0.0f;
    for (int j = lane; j < S; j += 32)
      s += __expf(alpha[(b << LOG2S) + j] + finals[j] - m);
    s = wave_sum32(s);
    if (lane == 0) redb[b] = m + __logf(s);
  }
  __syncthreads();
  if (tid == 0) {
    float tot = 0.0f;
#pragma unroll
    for (int b = 0; b < BATCH; ++b) tot += redb[b];
    *out_llh = tot;
  }
}

__global__ void lfmmi_finalize(const float* __restrict__ llh,
                               float* __restrict__ out) {
  // llh[0] = numerator total, llh[1] = denominator total, DEN_SCALE = 1.0
  out[0] = -(llh[0] - llh[1]);
}

extern "C" void kernel_launch(void* const* d_in, const int* in_sizes, int n_in,
                              void* d_out, int out_size, void* d_ws, size_t ws_size,
                              hipStream_t stream) {
  (void)in_sizes; (void)n_in; (void)out_size; (void)ws_size;

  const float* input     = (const float*)d_in[0];
  const int*   seqlen    = (const int*)d_in[1];
  const float* num_init  = (const float*)d_in[2];
  const float* num_trans = (const float*)d_in[3];
  const float* num_final = (const float*)d_in[4];
  const int*   num_pdf   = (const int*)d_in[5];
  const float* den_init  = (const float*)d_in[6];
  const float* den_trans = (const float*)d_in[7];
  const float* den_final = (const float*)d_in[8];
  const int*   den_pdf   = (const int*)d_in[9];

  // Workspace layout: [den Bpack 2MB][num Bpack 128KB][llh 2 floats]
  char* ws = (char*)d_ws;
  _Float16* Bden = (_Float16*)ws;
  _Float16* Bnum = (_Float16*)(ws + (size_t)1024 * 1024 * 2);
  float*    llh  = (float*)(ws + (size_t)1024 * 1024 * 2 + (size_t)256 * 256 * 2);

  // Pack exp(trans) into WMMA B-operand layout (runs every call; deterministic).
  lfmmi_pack_trans<1024, 10><<<(1024 * 1024) / 256, 256, 0, stream>>>(den_trans, Bden);
  lfmmi_pack_trans<256, 8><<<(256 * 256) / 256, 256, 0, stream>>>(num_trans, Bnum);

  const size_t lds_den = (size_t)BATCH * 1024 * 6 + 128;  // 96KB+ : alpha f32 + A f16 + reduce
  const size_t lds_num = (size_t)BATCH * 256 * 6 + 128;

  // CDNA5 WGP has 320KB LDS; raise the dynamic-LDS cap for the den kernel.
  (void)hipFuncSetAttribute((const void*)lfmmi_forward<1024, 10, 4>,
                            hipFuncAttributeMaxDynamicSharedMemorySize,
                            (int)lds_den);
  (void)hipFuncSetAttribute((const void*)lfmmi_forward<256, 8, 1>,
                            hipFuncAttributeMaxDynamicSharedMemorySize,
                            (int)lds_num);

  // Independent 1-block kernels (num & den FSMs) can overlap on the device.
  lfmmi_forward<1024, 10, 4><<<1, 512, lds_den, stream>>>(
      input, seqlen, den_init, Bden, den_final, den_pdf, &llh[1]);
  lfmmi_forward<256, 8, 1><<<1, 512, lds_num, stream>>>(
      input, seqlen, num_init, Bnum, num_final, num_pdf, &llh[0]);

  lfmmi_finalize<<<1, 1, 0, stream>>>(llh, (float*)d_out);
}
